// Attention_layer_83451214561623
// MI455X (gfx1250) — matrix operand order repaired and checked by reading, not compiled, not run on hardware
//
#include <hip/hip_runtime.h>

// ---------------------------------------------------------------------------
// Fused encoder self-attention for MI455X (gfx1250), bf16 WMMA path with
// double-buffered async global->LDS staging.
//   B=4, S=2048, D=1024, H=8, head_dim=128
// Reference quirk: the mask is added as a per-query constant BEFORE the
// softmax over keys -> softmax is invariant to it -> mathematically a no-op,
// so masks (and the unused encoder_output) are ignored.
// ---------------------------------------------------------------------------

#define D_MODEL   1024
#define N_HEADS   8
#define HEAD_DIM  128
#define BATCH     4
#define SEQ       2048
#define M_TOKENS  (BATCH * SEQ)   // 8192

typedef __attribute__((ext_vector_type(16))) __bf16 v16bf;
typedef __attribute__((ext_vector_type(8)))  float  v8f;
typedef __attribute__((ext_vector_type(4)))  int    v4i;

union FragBF { unsigned int u[8]; v16bf v; };

// --------------------------- async LDS staging -----------------------------
#if defined(__HIP_DEVICE_COMPILE__) && defined(__gfx1250__) && \
    __has_builtin(__builtin_amdgcn_global_load_async_to_lds_b128)
#define ASYNC_LDS 1
#else
#define ASYNC_LDS 0
#endif

#define AS_GLOBAL __attribute__((address_space(1)))
#define AS_LOCAL  __attribute__((address_space(3)))

// Copy 16 bytes global -> LDS. Async path uses the CDNA5
// GLOBAL_LOAD_ASYNC_TO_LDS_B128 (tracked by ASYNCcnt); fallback is a
// synchronous b128 load + ds_store_b128 with identical barrier structure.
__device__ __forceinline__ void stage16(const unsigned short* __restrict__ g,
                                        unsigned short* __restrict__ l) {
#if ASYNC_LDS
  __builtin_amdgcn_global_load_async_to_lds_b128(
      (AS_GLOBAL v4i*)g, (AS_LOCAL v4i*)l, /*offset=*/0, /*cpol=*/0);
#else
  *reinterpret_cast<uint4*>(l) = *reinterpret_cast<const uint4*>(g);
#endif
}

// Wait until at most N async loads remain outstanding. Async loads complete
// in order, so waiting down to the just-issued count guarantees all older
// loads (the previous tile) have landed in LDS.
template <int N>
__device__ __forceinline__ void wait_async() {
#if ASYNC_LDS
#if __has_builtin(__builtin_amdgcn_s_wait_asynccnt)
  __builtin_amdgcn_s_wait_asynccnt(N);
#else
  asm volatile("s_wait_asynccnt %0" ::"i"(N) : "memory");
#endif
#endif
}

// ------------------------------ bf16 helpers -------------------------------
__device__ __forceinline__ unsigned short f32_to_bf16(float f) {
  unsigned int u = __float_as_uint(f);
  unsigned int r = u + 0x7FFFu + ((u >> 16) & 1u);   // round-to-nearest-even
  return (unsigned short)(r >> 16);
}

// A-matrix fragment (16 rows x 32 k, bf16), CDNA5 WMMA layout (ISA 7.12.2).
// Source: row-major ushort matrix, k contiguous within a row.
// Lane L: row = L&15; half = L>>4; VGPR v holds K pair at
//   k0 = ((v>>2)<<4) + (half<<3) + ((v&3)<<1)
__device__ __forceinline__ void load_frag_a(FragBF& f,
                                            const unsigned short* __restrict__ base,
                                            int row_stride, int lane) {
  const int r  = lane & 15;
  const int hf = lane >> 4;
  const unsigned short* p = base + (size_t)r * row_stride;
#pragma unroll
  for (int v = 0; v < 8; ++v) {
    const int k0 = ((v >> 2) << 4) + (hf << 3) + ((v & 3) << 1);
    f.u[v] = *(const unsigned int*)(p + k0);
  }
}

// B-matrix fragment (32 k x 16 cols, bf16), CDNA5 WMMA layout.
// Source: B^T stored row-major (rows = output column n, k contiguous).
// Lane L: col = L&15; half = L>>4; VGPR v holds K pair at
//   k0 = ((v>>2)<<4) + (half<<3) + ((v&3)<<1)  (same order as A).
__device__ __forceinline__ void load_frag_b(FragBF& f,
                                            const unsigned short* __restrict__ base,
                                            int row_stride, int lane) {
  const int n  = lane & 15;
  const int hf = lane >> 4;
  const unsigned short* p = base + (size_t)n * row_stride;
#pragma unroll
  for (int v = 0; v < 8; ++v) {
    const int k0 = ((v >> 2) << 4) + (hf << 3) + ((v & 3) << 1);
    f.u[v] = *(const unsigned int*)(p + k0);
  }
}

__device__ __forceinline__ v8f wmma_bf16(const FragBF& a, const FragBF& b, v8f c) {
  return __builtin_amdgcn_wmma_f32_16x16x32_bf16(
      /*neg_a=*/false, a.v, /*neg_b=*/false, b.v,
      /*c_mod=*/(short)0, c, /*reuse_a=*/false, /*reuse_b=*/false);
}

// ---------------------------------------------------------------------------
// Kernel 1a: X (f32) -> Xbf (bf16), elementwise.
// ---------------------------------------------------------------------------
__global__ void cvt_x_bf16(const float* __restrict__ x,
                           unsigned short* __restrict__ xb, int n) {
  int i = blockIdx.x * blockDim.x + threadIdx.x;
  if (i < n) xb[i] = f32_to_bf16(x[i]);
}

// ---------------------------------------------------------------------------
// Kernel 1b: W (f32, KxN) -> W^T (bf16, NxK), for Q/K/V selected by blockIdx.z.
// ---------------------------------------------------------------------------
__global__ void cvt_wT_bf16(const float* __restrict__ wq,
                            const float* __restrict__ wk,
                            const float* __restrict__ wv,
                            unsigned short* __restrict__ wt) {
  const float* w = (blockIdx.z == 0) ? wq : (blockIdx.z == 1) ? wk : wv;
  unsigned short* o = wt + (size_t)blockIdx.z * D_MODEL * D_MODEL;
  int i = blockIdx.x * blockDim.x + threadIdx.x;       // i = n*D + k
  if (i < D_MODEL * D_MODEL) {
    int n = i >> 10;
    int k = i & (D_MODEL - 1);
    o[i] = f32_to_bf16(w[(size_t)k * D_MODEL + n]);
  }
}

// ---------------------------------------------------------------------------
// Kernel 2: QKV projection GEMM via WMMA + double-buffered LDS staging.
//   C[m, n] = sum_k Xbf[m, k] * W[k, n],  m in [0, 8192), n in [0, 1024)
// blockIdx.z selects Q/K/V. Block = 128 threads = 4 waves; block tile 64x64,
// wave tile 32x32 (2x2 WMMA accumulators), K stepped by 32. Per k-step, the
// block cooperatively stages a 64x32 X tile and 64x32 W^T tile into LDS
// (async, ping-pong buffers), so each tile is fetched from global once
// instead of twice per wave pair.
// Head split (faithful to reference reshape): n = c*H + h -> h = n&7, c = n>>3.
// Q/K written as (B,H,S,128) row-major; V written transposed as (B,H,128,S).
// Q is pre-scaled by 1/sqrt(head_dim).
// ---------------------------------------------------------------------------
__global__ void __launch_bounds__(128)
proj_qkv_wmma(const unsigned short* __restrict__ Xb,
              const unsigned short* __restrict__ Wt,
              unsigned short* __restrict__ Qo,
              unsigned short* __restrict__ Ko,
              unsigned short* __restrict__ Vt) {
  __shared__ __align__(16) unsigned short Xs[2][64][32];
  __shared__ __align__(16) unsigned short Ws[2][64][32];

  const int tid  = threadIdx.x;
  const int lane = tid & 31;
  const int wave = tid >> 5;
  const int sel  = blockIdx.z;
  const int bm0  = blockIdx.x * 64;
  const int bn0  = blockIdx.y * 64;
  const int wm   = (wave >> 1) * 32;
  const int wn   = (wave & 1) * 32;
  const unsigned short* W = Wt + (size_t)sel * D_MODEL * D_MODEL;

  // Stage one 64x32 X tile + 64x32 W^T tile (4 async b128 per thread).
  auto stage_tiles = [&](int kk, int buf) {
#pragma unroll
    for (int i = 0; i < 2; ++i) {
      const int c   = tid + i * 128;     // 0..255 chunk id (16B chunks)
      const int row = c >> 2;
      const int sub = (c & 3) << 3;      // ushort offset within row
      stage16(Xb + (size_t)(bm0 + row) * D_MODEL + kk + sub, &Xs[buf][row][sub]);
      stage16(W  + (size_t)(bn0 + row) * D_MODEL + kk + sub, &Ws[buf][row][sub]);
    }
  };

  stage_tiles(0, 0);

  v8f acc00 = {0,0,0,0,0,0,0,0}, acc01 = acc00, acc10 = acc00, acc11 = acc00;

  for (int kt = 0; kt < D_MODEL / 32; ++kt) {
    const int buf = kt & 1;
    if (kt + 1 < D_MODEL / 32) {
      stage_tiles((kt + 1) * 32, buf ^ 1);
      wait_async<4>();          // previous tile's 4 loads complete
    } else {
      wait_async<0>();
    }
    __syncthreads();            // all waves' stages for tile kt visible

    FragBF a0, a1, b0, b1;
    load_frag_a(a0, &Xs[buf][wm][0],      32, lane);
    load_frag_a(a1, &Xs[buf][wm + 16][0], 32, lane);
    load_frag_b(b0, &Ws[buf][wn][0],      32, lane);
    load_frag_b(b1, &Ws[buf][wn + 16][0], 32, lane);
    acc00 = wmma_bf16(a0, b0, acc00);
    acc01 = wmma_bf16(a0, b1, acc01);
    acc10 = wmma_bf16(a1, b0, acc10);
    acc11 = wmma_bf16(a1, b1, acc11);

    __syncthreads();            // everyone done reading buf before reuse
  }

  const float qscale = 0.088388347648318447f;  // 1/sqrt(128)
  const int mrow = ((lane >> 4) << 3);
  const int ncol = lane & 15;
#pragma unroll
  for (int i = 0; i < 2; ++i) {
#pragma unroll
    for (int j = 0; j < 2; ++j) {
      const v8f& acc = (i == 0) ? (j == 0 ? acc00 : acc01)
                                : (j == 0 ? acc10 : acc11);
#pragma unroll
      for (int v = 0; v < 8; ++v) {
        int m = bm0 + wm + i * 16 + mrow + v;
        int n = bn0 + wn + j * 16 + ncol;
        int bb = m >> 11;               // m / SEQ
        int ss = m & (SEQ - 1);
        int hh = n & (N_HEADS - 1);
        int cc = n >> 3;                // n / N_HEADS
        float val = acc[v];
        if (sel == 0) {
          Qo[(((size_t)bb * N_HEADS + hh) * SEQ + ss) * HEAD_DIM + cc] =
              f32_to_bf16(val * qscale);
        } else if (sel == 1) {
          Ko[(((size_t)bb * N_HEADS + hh) * SEQ + ss) * HEAD_DIM + cc] =
              f32_to_bf16(val);
        } else {
          Vt[(((size_t)bb * N_HEADS + hh) * HEAD_DIM + cc) * SEQ + ss] =
              f32_to_bf16(val);
        }
      }
    }
  }
}

// ---------------------------------------------------------------------------
// Kernel 3: flash attention per (b, h). Block = 128 threads = 4 waves; each
// wave owns 16 query rows and a persistent 16x128 f32 accumulator. Keys are
// processed 32 at a time; the K tile (32x128) and V^T tile (128x32) are
// cooperatively staged into double-buffered LDS (async), shared by all 4
// waves (4x less global traffic than per-wave fetches):
//   scores:  8 x WMMA   (4 K-chunks x 2 key subtiles), operands from LDS
//   softmax: online, per-row stats via __shfl_xor over the 16 lanes that
//            share a C-layout row group
//   P:       written bf16 to a per-wave LDS patch, reloaded as an A-fragment
//   PV:      8 x WMMA into persistent accumulators, V operand from LDS
// Output merge faithful to reference: out[b, s, c*H + h] = O[q, c].
// ---------------------------------------------------------------------------
__global__ void __launch_bounds__(128)
flash_attn_wmma(const unsigned short* __restrict__ Q,
                const unsigned short* __restrict__ K,
                const unsigned short* __restrict__ Vt,
                float* __restrict__ out) {
  __shared__ __align__(16) unsigned short Ks[2][32][HEAD_DIM];  // keys x feat
  __shared__ __align__(16) unsigned short Vs[2][HEAD_DIM][32];  // feat x keys
  __shared__ unsigned short pshare[4][16][32];                  // per-wave P

  const int tid   = threadIdx.x;
  const int lane  = tid & 31;
  const int wave  = tid >> 5;
  const int b     = blockIdx.z;
  const int h     = blockIdx.y;
  const int qbase = blockIdx.x * 64 + wave * 16;

  const unsigned short* Qp = Q  + ((size_t)b * N_HEADS + h) * SEQ * HEAD_DIM;
  const unsigned short* Kp = K  + ((size_t)b * N_HEADS + h) * SEQ * HEAD_DIM;
  const unsigned short* Vp = Vt + ((size_t)b * N_HEADS + h) * HEAD_DIM * SEQ;

  // Stage 32 keys: K tile is one contiguous 8KB block; V^T tile is 128 rows
  // of 64B. 8 async b128 chunks per thread.
  auto stage_kv = [&](int j0, int buf) {
    unsigned short* kdst = &Ks[buf][0][0];
#pragma unroll
    for (int i = 0; i < 4; ++i) {
      const int c = tid + i * 128;       // 0..511 chunk id (16B chunks)
      stage16(Kp + (size_t)j0 * HEAD_DIM + c * 8, kdst + c * 8);
      const int row = c >> 2;
      const int sub = (c & 3) << 3;
      stage16(Vp + (size_t)row * SEQ + j0 + sub, &Vs[buf][row][sub]);
    }
  };

  // Q tile (16 x 128) as 4 A-fragments, resident for the whole kernel.
  FragBF qf[4];
#pragma unroll
  for (int kc = 0; kc < 4; ++kc)
    load_frag_a(qf[kc], Qp + (size_t)qbase * HEAD_DIM + kc * 32, HEAD_DIM, lane);

  float mrow[8], lrow[8];
  v8f o[8];
#pragma unroll
  for (int v = 0; v < 8; ++v) { mrow[v] = -1e30f; lrow[v] = 0.0f; }
  {
    v8f z = {0,0,0,0,0,0,0,0};
#pragma unroll
    for (int t = 0; t < 8; ++t) o[t] = z;
  }

  stage_kv(0, 0);

  for (int jt = 0; jt < SEQ / 32; ++jt) {
    const int buf = jt & 1;
    if (jt + 1 < SEQ / 32) {
      stage_kv((jt + 1) * 32, buf ^ 1);
      wait_async<8>();          // tile jt's 8 loads complete (in-order)
    } else {
      wait_async<0>();
    }
    __syncthreads();            // all waves' stages for tile jt visible

    // ---- scores: S = Q * K^T for 32 keys (two 16x16 C tiles) ----
    v8f sc0 = {0,0,0,0,0,0,0,0};
    v8f sc1 = sc0;
#pragma unroll
    for (int kc = 0; kc < 4; ++kc) {
      FragBF kb0, kb1;
      load_frag_b(kb0, &Ks[buf][0][kc * 32],  HEAD_DIM, lane);
      load_frag_b(kb1, &Ks[buf][16][kc * 32], HEAD_DIM, lane);
      sc0 = wmma_bf16(qf[kc], kb0, sc0);
      sc1 = wmma_bf16(qf[kc], kb1, sc1);
    }

    // ---- online softmax over the 32 new columns ----
    float alpha[8];
#pragma unroll
    for (int v = 0; v < 8; ++v) {
      float cm = fmaxf(sc0[v], sc1[v]);
      cm = fmaxf(cm, __shfl_xor(cm, 1, 32));
      cm = fmaxf(cm, __shfl_xor(cm, 2, 32));
      cm = fmaxf(cm, __shfl_xor(cm, 4, 32));
      cm = fmaxf(cm, __shfl_xor(cm, 8, 32));
      float nm = fmaxf(mrow[v], cm);
      alpha[v] = __expf(mrow[v] - nm);
      mrow[v]  = nm;

      float p0 = __expf(sc0[v] - nm);
      float p1 = __expf(sc1[v] - nm);
      const int m = ((lane >> 4) << 3) + v;
      const int n = lane & 15;
      pshare[wave][m][n]      = f32_to_bf16(p0);
      pshare[wave][m][16 + n] = f32_to_bf16(p1);

      float cs = p0 + p1;
      cs += __shfl_xor(cs, 1, 32);
      cs += __shfl_xor(cs, 2, 32);
      cs += __shfl_xor(cs, 4, 32);
      cs += __shfl_xor(cs, 8, 32);
      lrow[v] = lrow[v] * alpha[v] + cs;
    }

    // ---- P (16x32 bf16) back out of LDS in A-fragment layout ----
    FragBF pf;
    load_frag_a(pf, &pshare[wave][0][0], 32, lane);

    // ---- O = O*alpha + P * V  (8 feature chunks of 16) ----
#pragma unroll
    for (int t = 0; t < 8; ++t) {
#pragma unroll
      for (int v = 0; v < 8; ++v) o[t][v] *= alpha[v];
      FragBF vb;
      load_frag_b(vb, &Vs[buf][t * 16][0], 32, lane);
      o[t] = wmma_bf16(pf, vb, o[t]);
    }

    __syncthreads();            // everyone done reading buf before reuse
  }

  // ---- normalize and store (out[b, s, c*H + h]) ----
#pragma unroll
  for (int t = 0; t < 8; ++t) {
#pragma unroll
    for (int v = 0; v < 8; ++v) {
      int m = ((lane >> 4) << 3) + v;
      int s = qbase + m;
      int c = t * 16 + (lane & 15);
      int d = c * N_HEADS + h;
      out[((size_t)b * SEQ + s) * D_MODEL + d] = o[t][v] / lrow[v];
    }
  }
}

// ---------------------------------------------------------------------------
// Launcher. Workspace layout (bytes):
//   [0,16M)   Xbf    : 8192 x 1024 bf16
//   [16M,22M) Wt     : 3 x (1024 x 1024) bf16, transposed (n-major)
//   [22M,38M) Q      : (B,H,S,128) bf16, pre-scaled by 1/sqrt(128)
//   [38M,54M) K      : (B,H,S,128) bf16
//   [54M,70M) V^T    : (B,H,128,S) bf16
// ---------------------------------------------------------------------------
extern "C" void kernel_launch(void* const* d_in, const int* in_sizes, int n_in,
                              void* d_out, int out_size, void* d_ws, size_t ws_size,
                              hipStream_t stream) {
  const float* x  = (const float*)d_in[0];
  // d_in[1] encoder_output_embedding: unused (attention_type == 'encoder')
  const float* wq = (const float*)d_in[2];
  const float* wk = (const float*)d_in[3];
  const float* wv = (const float*)d_in[4];
  // d_in[5], d_in[6] masks: mathematically a softmax no-op (see header comment)
  float* out = (float*)d_out;

  char* ws = (char*)d_ws;
  const size_t MB = 1024 * 1024;
  unsigned short* Xb = (unsigned short*)(ws);
  unsigned short* Wt = (unsigned short*)(ws + 16 * MB);
  unsigned short* Qw = (unsigned short*)(ws + 22 * MB);
  unsigned short* Kw = (unsigned short*)(ws + 38 * MB);
  unsigned short* Vt = (unsigned short*)(ws + 54 * MB);

  {
    int n = M_TOKENS * D_MODEL;
    cvt_x_bf16<<<(n + 255) / 256, 256, 0, stream>>>(x, Xb, n);
  }
  {
    dim3 g((D_MODEL * D_MODEL + 255) / 256, 1, 3);
    cvt_wT_bf16<<<g, 256, 0, stream>>>(wq, wk, wv, Wt);
  }
  {
    dim3 g(M_TOKENS / 64, D_MODEL / 64, 3);   // 128 x 16 x 3
    proj_qkv_wmma<<<g, 128, 0, stream>>>(Xb, Wt, Qw, Kw, Vt);
  }
  {
    dim3 g(SEQ / 64, N_HEADS, BATCH);          // 32 x 8 x 4
    flash_attn_wmma<<<g, 128, 0, stream>>>(Qw, Kw, Vt, out);
  }
}
